// WaveFormer_7816840479020
// MI455X (gfx1250) — matrix operand order repaired and checked by reading, not compiled, not run on hardware
//
#include <hip/hip_runtime.h>
#include <math.h>

// ---------------------------------------------------------------------------
// WaveFormer forward for MI455X (gfx1250), wave32 + WMMA f16 (f32 accum).
// All matmuls -> v_wmma_f32_16x16x32_f16. Attention keeps the probability
// strip in LDS (f16) so the P*V WMMA reads A-fragments via ds loads.
// ---------------------------------------------------------------------------

typedef _Float16 v16h __attribute__((ext_vector_type(16)));
typedef _Float16 v8h  __attribute__((ext_vector_type(8)));
typedef float    v8f  __attribute__((ext_vector_type(8)));

#define NB      1024
#define TT      32
#define DMODEL  256
#define NTOK    (NB * TT)        // 32768
#define KPROJ   160              // 158 padded to multiple of 32-lane chunks
#define DFEAT   158
#define GDIM    63

__device__ __forceinline__ v8f wmma16(v16h a, v16h b, v8f c) {
    return __builtin_amdgcn_wmma_f32_16x16x32_f16(
        /*neg_a=*/false, a, /*neg_b=*/false, b,
        /*c_mod=*/(short)0, c, /*reuse_a=*/false, /*reuse_b=*/false);
}

// --------------------------- weight conversion -----------------------------
__global__ void f32_to_f16_k(const float* __restrict__ s, _Float16* __restrict__ d, int n) {
    int i = blockIdx.x * blockDim.x + threadIdx.x;
    if (i < n) d[i] = (_Float16)s[i];
}

// proj_w (256 x 158) -> f16 padded to (256 x 160)
__global__ void proj_pad_k(const float* __restrict__ s, _Float16* __restrict__ d) {
    int i = blockIdx.x * blockDim.x + threadIdx.x;
    if (i < DMODEL * KPROJ) {
        int o = i / KPROJ, k = i % KPROJ;
        d[i] = (k < DFEAT) ? (_Float16)s[o * DFEAT + k] : (_Float16)0.0f;
    }
}

// ------------------------------ gate + scale -------------------------------
// gate = 158 * softmax((x[n,-1,158:221] @ gate_w.T + gate_b)/5); src16 = f16(x*gate), K padded to 160
__global__ void gate_src_k(const float* __restrict__ x, const float* __restrict__ gw,
                           const float* __restrict__ gb, _Float16* __restrict__ src16) {
    __shared__ float lg[DFEAT];
    __shared__ float gsh[KPROJ];
    __shared__ float red[2];
    int n = blockIdx.x, tid = threadIdx.x;
    const float* gin = x + ((size_t)(n * TT + (TT - 1)) * 221 + DFEAT);
    if (tid < DFEAT) {
        float s = gb[tid];
        for (int i = 0; i < GDIM; ++i) s += gin[i] * gw[tid * GDIM + i];
        lg[tid] = s * 0.2f;  // / BETA
    }
    __syncthreads();
    if (tid == 0) {
        float m = -3.0e38f;
        for (int j = 0; j < DFEAT; ++j) m = fmaxf(m, lg[j]);
        float ss = 0.f;
        for (int j = 0; j < DFEAT; ++j) ss += __expf(lg[j] - m);
        red[0] = m; red[1] = ss;
    }
    __syncthreads();
    if (tid < KPROJ)
        gsh[tid] = (tid < DFEAT) ? (float)DFEAT * __expf(lg[tid] - red[0]) / red[1] : 0.f;
    __syncthreads();
    for (int idx = tid; idx < TT * KPROJ; idx += blockDim.x) {
        int t = idx / KPROJ, f = idx % KPROJ;
        float v = (f < DFEAT) ? x[(size_t)(n * TT + t) * 221 + f] * gsh[f] : 0.f;
        src16[(size_t)(n * TT + t) * KPROJ + f] = (_Float16)v;
    }
}

// --------------------------- generic WMMA GEMM -----------------------------
// C[M x Nout] = act16[M x K] @ W16[Nout x K]^T (+bias)(ReLU)(+resid), out f32/f16
__global__ __launch_bounds__(256) void gemm_k(
    const _Float16* __restrict__ Aact, int lda,
    const _Float16* __restrict__ W, int ldw,
    const float* __restrict__ bias, const float* __restrict__ resid,
    float* __restrict__ outF, _Float16* __restrict__ outH,
    int ldc, int M, int Nout, int K, int relu) {
    int wave = threadIdx.x >> 5, lane = threadIdx.x & 31;
    int half = lane >> 4, lr = lane & 15;
    int tilesN = Nout >> 4;
    int tid = blockIdx.x * 8 + wave;
    int total = (M >> 4) * tilesN;
    if (tid >= total) return;  // wave-uniform
    int tm = (tid / tilesN) << 4, tn = (tid % tilesN) << 4;
    const _Float16* ar = Aact + (size_t)(tm + lr) * lda;
    const _Float16* br = W + (size_t)(tn + lr) * ldw;
    v8f acc = {};
    for (int k0 = 0; k0 < K; k0 += 32) {
        __builtin_prefetch(ar + k0 + 64, 0, 1);
        v8h alo = *(const v8h*)(ar + k0 + half * 8);
        v8h ahi = *(const v8h*)(ar + k0 + 16 + half * 8);
        v8h b0  = *(const v8h*)(br + k0 + half * 8);
        v8h b1  = *(const v8h*)(br + k0 + 16 + half * 8);
        v16h a, b;
#pragma unroll
        for (int i = 0; i < 8; ++i) { a[i] = alo[i]; a[8 + i] = ahi[i]; b[i] = b0[i]; b[8 + i] = b1[i]; }
        acc = wmma16(a, b, acc);
    }
#pragma unroll
    for (int v = 0; v < 8; ++v) {
        int r = tm + half * 8 + v, c = tn + lr;
        float val = acc[v];
        if (bias)  val += bias[c];
        if (relu)  val = fmaxf(val, 0.f);
        if (resid) val += resid[(size_t)r * ldc + c];
        if (outF)  outF[(size_t)r * ldc + c] = val;
        if (outH)  outH[(size_t)r * ldc + c] = (_Float16)val;
    }
}

// ------------------------------ positional enc -----------------------------
__global__ void posenc_k(float* __restrict__ A) {
    int idx = blockIdx.x * blockDim.x + threadIdx.x;  // N*T*256 elements
    int d = idx & 255;
    int t = (idx >> 8) & 31;
    float de = (float)(d & ~1);
    float freq = __expf(de * (-9.210340371976184f / 256.f));  // -ln(10000)/d_model * (2p)
    float ang = (float)t * freq;
    A[idx] += (d & 1) ? __cosf(ang) : __sinf(ang);
}

// -------------------------------- LayerNorm --------------------------------
// wave-per-token; writes f32 (optional in-place) + f16 mirror
__global__ __launch_bounds__(256) void ln_k(const float* __restrict__ in,
                                            float* __restrict__ outF, _Float16* __restrict__ outH,
                                            const float* __restrict__ g, const float* __restrict__ b) {
    int wave = threadIdx.x >> 5, lane = threadIdx.x & 31;
    int tok = blockIdx.x * 8 + wave;
    const float* p = in + (size_t)tok * DMODEL;
    float v[8];
    float s = 0.f;
#pragma unroll
    for (int i = 0; i < 8; ++i) { v[i] = p[lane + 32 * i]; s += v[i]; }
#pragma unroll
    for (int o = 16; o; o >>= 1) s += __shfl_xor(s, o, 32);
    float mu = s * (1.f / 256.f);
    float q = 0.f;
#pragma unroll
    for (int i = 0; i < 8; ++i) { float d = v[i] - mu; q += d * d; }
#pragma unroll
    for (int o = 16; o; o >>= 1) q += __shfl_xor(q, o, 32);
    float r = rsqrtf(q * (1.f / 256.f) + 1e-5f);
#pragma unroll
    for (int i = 0; i < 8; ++i) {
        int d = lane + 32 * i;
        float o = (v[i] - mu) * r * g[d] + b[d];
        if (outF) outF[(size_t)tok * DMODEL + d] = o;
        outH[(size_t)tok * DMODEL + d] = (_Float16)o;
    }
}

// -------------------------------- attention --------------------------------
// One wave per (seq, head, 16-query strip). token_row = seq*SEQA + pos*POSA.
// Stage1: scores strip (16 x S) via WMMA -> f16 in LDS. Softmax in-wave.
// Stage2: P(16 x S) @ V(S x DH) via WMMA (A from LDS). out = resid + attn.
template <int S, int DH, int SEQA, int POSA>
__global__ __launch_bounds__(32) void attn_k(const _Float16* __restrict__ Q,
                                             const _Float16* __restrict__ Km,
                                             const _Float16* __restrict__ V,
                                             const float* __restrict__ resid,
                                             float* __restrict__ outF,
                                             int nhead, float invTemp) {
    extern __shared__ _Float16 pp[];  // S*16 halfs
    int lane = threadIdx.x & 31, half = lane >> 4, lr = lane & 15;
    constexpr int QT = S >> 4;
    int bid = blockIdx.x;
    int qt = bid % QT;
    int tmp = bid / QT;
    int head = tmp % nhead;
    int seq = tmp / nhead;
    int qbase = qt << 4;
    constexpr int NKC = DH >> 5;

    // preload Q fragments (row M = query lr of strip)
    v16h qa[NKC];
    const _Float16* qrow = Q + (size_t)(seq * SEQA + (qbase + lr) * POSA) * DMODEL + head * DH;
#pragma unroll
    for (int kc = 0; kc < NKC; ++kc) {
        int k0 = kc << 5;
        v8h lo = *(const v8h*)(qrow + k0 + half * 8);
        v8h hi = *(const v8h*)(qrow + k0 + 16 + half * 8);
#pragma unroll
        for (int i = 0; i < 8; ++i) { qa[kc][i] = lo[i]; qa[kc][8 + i] = hi[i]; }
    }

    // ---- stage 1: scores ----
    for (int ct = 0; ct < QT; ++ct) {
        const _Float16* krow = Km + (size_t)(seq * SEQA + (ct * 16 + lr) * POSA) * DMODEL + head * DH;
        v8f acc = {};
#pragma unroll
        for (int kc = 0; kc < NKC; ++kc) {
            int k0 = kc << 5;
            v8h b0 = *(const v8h*)(krow + k0 + half * 8);
            v8h b1 = *(const v8h*)(krow + k0 + 16 + half * 8);
            v16h b;
#pragma unroll
            for (int i = 0; i < 8; ++i) { b[i] = b0[i]; b[8 + i] = b1[i]; }
            acc = wmma16(qa[kc], b, acc);
        }
#pragma unroll
        for (int v = 0; v < 8; ++v)
            pp[(half * 8 + v) * S + ct * 16 + lr] = (_Float16)(acc[v] * invTemp);
    }
    __syncthreads();

    // ---- softmax over each of the 16 rows ----
    constexpr int PER = S >> 5;
    for (int r = 0; r < 16; ++r) {
        float m = -3.0e38f;
#pragma unroll
        for (int i = 0; i < PER; ++i) m = fmaxf(m, (float)pp[r * S + lane + 32 * i]);
#pragma unroll
        for (int o = 16; o; o >>= 1) m = fmaxf(m, __shfl_xor(m, o, 32));
        float s = 0.f;
        float e[PER];
#pragma unroll
        for (int i = 0; i < PER; ++i) { e[i] = __expf((float)pp[r * S + lane + 32 * i] - m); s += e[i]; }
#pragma unroll
        for (int o = 16; o; o >>= 1) s += __shfl_xor(s, o, 32);
        float inv = 1.f / s;
#pragma unroll
        for (int i = 0; i < PER; ++i) pp[r * S + lane + 32 * i] = (_Float16)(e[i] * inv);
    }
    __syncthreads();

    // ---- stage 2: P @ V ----
    constexpr int DTILES = DH >> 4;
#pragma unroll
    for (int dt = 0; dt < DTILES; ++dt) {
        v8f acc = {};
        for (int k0 = 0; k0 < S; k0 += 32) {
            v8h lo = *(const v8h*)(pp + lr * S + k0 + half * 8);   // ds loads
            v8h hi = *(const v8h*)(pp + lr * S + k0 + 16 + half * 8);
            v16h a, b;
#pragma unroll
            for (int i = 0; i < 8; ++i) { a[i] = lo[i]; a[8 + i] = hi[i]; }
#pragma unroll
            for (int j = 0; j < 16; ++j) {
                int kk = k0 + half * 8 + (j & 7) + ((j >> 3) << 4);
                b[j] = V[(size_t)(seq * SEQA + kk * POSA) * DMODEL + head * DH + dt * 16 + lr];
            }
            acc = wmma16(a, b, acc);
        }
#pragma unroll
        for (int v = 0; v < 8; ++v) {
            int qi = qbase + half * 8 + v;
            size_t row = (size_t)(seq * SEQA + qi * POSA);
            int col = head * DH + dt * 16 + lr;
            outF[row * DMODEL + col] = resid[row * DMODEL + col] + acc[v];
        }
    }
}

// ------------------------- pooling + output head ---------------------------
__global__ __launch_bounds__(256) void pool_k(const float* __restrict__ h, const float* __restrict__ z,
                                              const float* __restrict__ ow, const float* __restrict__ ob,
                                              float* __restrict__ out) {
    __shared__ float red[256];
    __shared__ float lam[TT];
    __shared__ float hl[DMODEL];
    int n = blockIdx.x, tid = threadIdx.x;
    hl[tid] = h[(size_t)(n * TT + (TT - 1)) * DMODEL + tid];
    __syncthreads();
    for (int t = 0; t < TT; ++t) {
        red[tid] = h[(size_t)(n * TT + t) * DMODEL + tid] * hl[tid];
        __syncthreads();
        for (int o = 128; o; o >>= 1) { if (tid < o) red[tid] += red[tid + o]; __syncthreads(); }
        if (tid == 0) lam[t] = red[0];
        __syncthreads();
    }
    if (tid == 0) {
        float m = -3.0e38f;
        for (int t = 0; t < TT; ++t) m = fmaxf(m, lam[t]);
        float s = 0.f;
        for (int t = 0; t < TT; ++t) { lam[t] = __expf(lam[t] - m); s += lam[t]; }
        float inv = 1.f / s;
        for (int t = 0; t < TT; ++t) lam[t] *= inv;
    }
    __syncthreads();
    float p = 0.f;
    for (int t = 0; t < TT; ++t) p += lam[t] * z[(size_t)(n * TT + t) * DMODEL + tid];
    red[tid] = p * ow[tid];
    __syncthreads();
    for (int o = 128; o; o >>= 1) { if (tid < o) red[tid] += red[tid + o]; __syncthreads(); }
    if (tid == 0) out[n] = red[0] + ob[0];
}

// ---------------------------------------------------------------------------
extern "C" void kernel_launch(void* const* d_in, const int* in_sizes, int n_in,
                              void* d_out, int out_size, void* d_ws, size_t ws_size,
                              hipStream_t stream) {
    (void)in_sizes; (void)n_in; (void)out_size; (void)ws_size;
    const float* x      = (const float*)d_in[0];
    const float* gate_w = (const float*)d_in[1];
    const float* gate_b = (const float*)d_in[2];
    const float* proj_w = (const float*)d_in[3];
    const float* proj_b = (const float*)d_in[4];
    const float* tn1g = (const float*)d_in[8],  *tn1b = (const float*)d_in[9];
    const float* tn2g = (const float*)d_in[10], *tn2b = (const float*)d_in[11];
    const float* tf1b = (const float*)d_in[13], *tf2b = (const float*)d_in[15];
    const float* sn1g = (const float*)d_in[19], *sn1b = (const float*)d_in[20];
    const float* sn2g = (const float*)d_in[21], *sn2b = (const float*)d_in[22];
    const float* sf1b = (const float*)d_in[24], *sf2b = (const float*)d_in[26];
    const float* out_w = (const float*)d_in[28], *out_b = (const float*)d_in[29];

    // bump allocator over d_ws
    char* ws = (char*)d_ws;
    size_t off = 0;
    auto alloc = [&](size_t bytes) -> char* {
        char* p = ws + off;
        off += (bytes + 255) & ~(size_t)255;
        return p;
    };
    _Float16* projw16 = (_Float16*)alloc((size_t)DMODEL * KPROJ * 2);
    const int widx[11] = {5, 6, 7, 12, 14, 16, 17, 18, 23, 25, 27};
    _Float16* w16[11];
    for (int i = 0; i < 11; ++i) w16[i] = (_Float16*)alloc((size_t)DMODEL * DMODEL * 2);
    _Float16* TQ = w16[0], *TK = w16[1], *TV = w16[2], *TF1 = w16[3], *TF2 = w16[4];
    _Float16* SQ = w16[5], *SK = w16[6], *SV = w16[7], *SF1 = w16[8], *SF2 = w16[9];
    _Float16* TEMPW = w16[10];
    _Float16* src16 = (_Float16*)alloc((size_t)NTOK * KPROJ * 2);
    float*    Af    = (float*)alloc((size_t)NTOK * DMODEL * 4);
    float*    Bf    = (float*)alloc((size_t)NTOK * DMODEL * 4);
    _Float16* H0    = (_Float16*)alloc((size_t)NTOK * DMODEL * 2);
    _Float16* H1    = (_Float16*)alloc((size_t)NTOK * DMODEL * 2);
    _Float16* H2    = (_Float16*)alloc((size_t)NTOK * DMODEL * 2);
    _Float16* H3    = (_Float16*)alloc((size_t)NTOK * DMODEL * 2);

    // --- weight conversion (f32 -> f16) ---
    proj_pad_k<<<(DMODEL * KPROJ + 255) / 256, 256, 0, stream>>>(proj_w, projw16);
    for (int i = 0; i < 11; ++i)
        f32_to_f16_k<<<(DMODEL * DMODEL + 255) / 256, 256, 0, stream>>>(
            (const float*)d_in[widx[i]], w16[i], DMODEL * DMODEL);

    // --- gate + scaled src (f16, K padded to 160) ---
    gate_src_k<<<NB, 256, 0, stream>>>(x, gate_w, gate_b, src16);

    const int GEMM_GRID = (NTOK / 16) * (DMODEL / 16) / 8;  // 4096
    // --- projection: A = src @ proj_w.T + proj_b ---
    gemm_k<<<GEMM_GRID, 256, 0, stream>>>(src16, KPROJ, projw16, KPROJ, proj_b, nullptr,
                                          Af, nullptr, DMODEL, NTOK, DMODEL, KPROJ, 0);
    posenc_k<<<NTOK * DMODEL / 256, 256, 0, stream>>>(Af);

    // ================= block 1 (time attention: seq=n, len 32) =============
    ln_k<<<NTOK / 8, 256, 0, stream>>>(Af, Bf, H0, tn1g, tn1b);  // xn (f32 in Bf, f16 in H0)
    gemm_k<<<GEMM_GRID, 256, 0, stream>>>(H0, DMODEL, TQ, DMODEL, nullptr, nullptr,
                                          nullptr, H1, DMODEL, NTOK, DMODEL, DMODEL, 0);
    gemm_k<<<GEMM_GRID, 256, 0, stream>>>(H0, DMODEL, TK, DMODEL, nullptr, nullptr,
                                          nullptr, H2, DMODEL, NTOK, DMODEL, DMODEL, 0);
    gemm_k<<<GEMM_GRID, 256, 0, stream>>>(H0, DMODEL, TV, DMODEL, nullptr, nullptr,
                                          nullptr, H3, DMODEL, NTOK, DMODEL, DMODEL, 0);
    {
        size_t sm = (size_t)32 * 16 * 2;
        attn_k<32, 64, TT, 1><<<NB * 4 * (32 / 16), 32, sm, stream>>>(H1, H2, H3, Bf, Af, 4, 1.0f);
    }
    ln_k<<<NTOK / 8, 256, 0, stream>>>(Af, Af, H0, tn2g, tn2b);  // xt
    gemm_k<<<GEMM_GRID, 256, 0, stream>>>(H0, DMODEL, TF1, DMODEL, tf1b, nullptr,
                                          nullptr, H1, DMODEL, NTOK, DMODEL, DMODEL, 1);
    gemm_k<<<GEMM_GRID, 256, 0, stream>>>(H1, DMODEL, TF2, DMODEL, tf2b, Af,
                                          Af, nullptr, DMODEL, NTOK, DMODEL, DMODEL, 0);

    // ================= block 2 (space attention: seq=t, len 1024) ==========
    ln_k<<<NTOK / 8, 256, 0, stream>>>(Af, Bf, H0, sn1g, sn1b);
    gemm_k<<<GEMM_GRID, 256, 0, stream>>>(H0, DMODEL, SQ, DMODEL, nullptr, nullptr,
                                          nullptr, H1, DMODEL, NTOK, DMODEL, DMODEL, 0);
    gemm_k<<<GEMM_GRID, 256, 0, stream>>>(H0, DMODEL, SK, DMODEL, nullptr, nullptr,
                                          nullptr, H2, DMODEL, NTOK, DMODEL, DMODEL, 0);
    gemm_k<<<GEMM_GRID, 256, 0, stream>>>(H0, DMODEL, SV, DMODEL, nullptr, nullptr,
                                          nullptr, H3, DMODEL, NTOK, DMODEL, DMODEL, 0);
    {
        size_t sm = (size_t)1024 * 16 * 2;  // 32 KB f16 prob strip
        attn_k<1024, 128, 1, TT><<<TT * 2 * (1024 / 16), 32, sm, stream>>>(
            H1, H2, H3, Bf, Af, 2, 0.08838834764831845f /* 1/sqrt(128) */);
    }
    ln_k<<<NTOK / 8, 256, 0, stream>>>(Af, Af, H0, sn2g, sn2b);
    gemm_k<<<GEMM_GRID, 256, 0, stream>>>(H0, DMODEL, SF1, DMODEL, sf1b, nullptr,
                                          nullptr, H1, DMODEL, NTOK, DMODEL, DMODEL, 1);
    // final residual add; also emit f16 mirror of z for the temp_w GEMM
    gemm_k<<<GEMM_GRID, 256, 0, stream>>>(H1, DMODEL, SF2, DMODEL, sf2b, Af,
                                          Af, H0, DMODEL, NTOK, DMODEL, DMODEL, 0);

    // ================= head: h = z @ temp_w.T, pooled softmax ==============
    gemm_k<<<GEMM_GRID, 256, 0, stream>>>(H0, DMODEL, TEMPW, DMODEL, nullptr, nullptr,
                                          Bf, nullptr, DMODEL, NTOK, DMODEL, DMODEL, 0);
    pool_k<<<NB, 256, 0, stream>>>(Bf, Af, out_w, out_b, (float*)d_out);
}
